// StreamingAttentionBlock_33054068310238
// MI455X (gfx1250) — matrix operand order repaired and checked by reading, not compiled, not run on hardware
//
#include <hip/hip_runtime.h>
#include <hip/hip_bf16.h>

// ---------------------------------------------------------------------------
// StreamingAttentionBlock for MI455X (gfx1250, wave32, WMMA + async-LDS DMA)
//   qkv = x @ Wqkv            (4096x1024 @ 1024x3072)  -> bf16 WMMA GEMM
//   banded attention (j in [i, i+2])                   -> wave32 VALU kernel
//   out = attn @ Wproj + b    (4096x1024 @ 1024x1024)  -> bf16 WMMA GEMM
// ---------------------------------------------------------------------------

typedef __attribute__((ext_vector_type(16))) __bf16 v16bf;
typedef __attribute__((ext_vector_type(8)))  float  v8f;

#define DIM     1024
#define NHEAD   16
#define HD      64
#define SEQ     2048
#define BATCH   2
#define MTOT    (BATCH * SEQ)     // 4096 rows
#define NQKV    (3 * DIM)         // 3072

// GEMM tiling: block 128(M) x 256(N), K-step 32; 8 waves, wave tile 64x64.
#define BM 128
#define BN 256
#define BK 32
#define LDP 8                       // LDS row pad (bf16) -> 16B, keeps b128 align
#define LDAK (BK + LDP)             // 40 bf16 per LDS row
#define A_TILE_SH (BM * LDAK)       // 5120 shorts / buffer
#define B_TILE_SH (BN * LDAK)       // 10240 shorts / buffer
#define A_BUF_BYTES (A_TILE_SH * 2) // 10240 B
#define B_BUF_BYTES (B_TILE_SH * 2) // 20480 B

static __device__ __forceinline__ unsigned short f32_to_bf16(float f) {
    unsigned int u = __float_as_uint(f);
    unsigned int r = (u + 0x7FFFu + ((u >> 16) & 1u)) >> 16;   // RNE
    return (unsigned short)r;
}

// CDNA5 async global->LDS copy (ASYNCcnt path), 16B per lane.
static __device__ __forceinline__ void async_copy_b128(unsigned lds_off,
                                                       const void* gptr) {
    asm volatile("global_load_async_to_lds_b128 %0, %1, off"
                 :: "v"(lds_off), "v"(gptr)
                 : "memory");
}

static __device__ __forceinline__ void wait_async0() {
    asm volatile("s_wait_asynccnt 0x0" ::: "memory");
}

// ---------------------------------------------------------------------------
// Cast fp32 -> bf16 (vectorized x4)
// ---------------------------------------------------------------------------
__global__ void cast_f32_bf16_kernel(const float* __restrict__ in,
                                     unsigned short* __restrict__ out, int n) {
    int i = (blockIdx.x * blockDim.x + threadIdx.x) * 4;
    if (i + 3 < n) {
        float4 v = *(const float4*)(in + i);
        unsigned int lo = ((unsigned int)f32_to_bf16(v.y) << 16) | f32_to_bf16(v.x);
        unsigned int hi = ((unsigned int)f32_to_bf16(v.w) << 16) | f32_to_bf16(v.z);
        uint2 p; p.x = lo; p.y = hi;
        *(uint2*)(out + i) = p;
    }
}

// ---------------------------------------------------------------------------
// Transpose + cast: W (K x N, fp32, row-major)  ->  WT (N x K, bf16, row-major)
// ---------------------------------------------------------------------------
__global__ void transpose_cast_kernel(const float* __restrict__ W,
                                      unsigned short* __restrict__ WT,
                                      int K, int N) {
    __shared__ float tile[32][33];
    int n0 = blockIdx.x * 32;
    int k0 = blockIdx.y * 32;
    int tx = threadIdx.x;          // 0..31
    int ty = threadIdx.y;          // 0..7
#pragma unroll
    for (int i = 0; i < 32; i += 8)
        tile[ty + i][tx] = W[(size_t)(k0 + ty + i) * N + n0 + tx];
    __syncthreads();
#pragma unroll
    for (int i = 0; i < 32; i += 8)
        WT[(size_t)(n0 + ty + i) * K + k0 + tx] = f32_to_bf16(tile[tx][ty + i]);
}

// ---------------------------------------------------------------------------
// bf16 WMMA GEMM:  C(MxN, f32) = A(MxK, bf16 row-major) * Bt(NxK, bf16 row-major)
// 256 threads = 8 waves (2 in M x 4 in N); wave tile 64x64 = 4x4 WMMA frags.
// Double-buffered LDS fed by global_load_async_to_lds_b128 (ASYNCcnt).
// ---------------------------------------------------------------------------
__global__ void __launch_bounds__(256)
gemm_bf16_wmma_kernel(const unsigned short* __restrict__ A,
                      const unsigned short* __restrict__ Bt,
                      float* __restrict__ C,
                      const float* __restrict__ bias,
                      int M, int N, int K, int has_bias) {
    __shared__ __align__(16) unsigned short As[2 * A_TILE_SH];
    __shared__ __align__(16) unsigned short Bs[2 * B_TILE_SH];

    const int tid    = threadIdx.x;
    const int lane   = tid & 31;
    const int wid    = tid >> 5;
    const int wave_m = wid & 1;        // 0..1  (64-row slab)
    const int wave_n = wid >> 1;       // 0..3  (64-col slab)
    const int m0     = blockIdx.y * BM;
    const int n0     = blockIdx.x * BN;

    const int lhalf = lane >> 4;       // 0 or 1
    const int lmod  = lane & 15;

    // Per-thread staging: A -> 2 chunks, B -> 4 chunks of 16B per K-tile.
    // Running global pointers (advanced by BK each tile) + buffer-0 LDS offsets.
    const unsigned short* aP[2];
    const unsigned short* bP[4];
    unsigned ldsA0[2], ldsB0[4];
#pragma unroll
    for (int c = 0; c < 2; ++c) {
        int lin = c * 2048 + tid * 8;  // A tile: 128x32 = 4096 shorts
        int row = lin >> 5, col = lin & 31;
        aP[c]    = A + (size_t)(m0 + row) * K + col;
        ldsA0[c] = (unsigned)(unsigned long long)(const void*)
                   &As[row * LDAK + col];
    }
#pragma unroll
    for (int c = 0; c < 4; ++c) {
        int lin = c * 2048 + tid * 8;  // B tile: 256x32 = 8192 shorts
        int row = lin >> 5, col = lin & 31;
        bP[c]    = Bt + (size_t)(n0 + row) * K + col;
        ldsB0[c] = (unsigned)(unsigned long long)(const void*)
                   &Bs[row * LDAK + col];
    }

    v8f acc[4][4];
#pragma unroll
    for (int f = 0; f < 4; ++f)
#pragma unroll
        for (int g = 0; g < 4; ++g)
            acc[f][g] = (v8f){0.f, 0.f, 0.f, 0.f, 0.f, 0.f, 0.f, 0.f};

    union FragU { uint4 u4[2]; v16bf v; };

    const int nk = K / BK;

    // prologue: DMA tile 0 into buffer 0, advance pointers to tile 1
#pragma unroll
    for (int c = 0; c < 2; ++c) { async_copy_b128(ldsA0[c], aP[c]); aP[c] += BK; }
#pragma unroll
    for (int c = 0; c < 4; ++c) { async_copy_b128(ldsB0[c], bP[c]); bP[c] += BK; }

    for (int kt = 0; kt < nk; ++kt) {
        wait_async0();          // this wave's share of tile kt is in LDS
        __syncthreads();        // everyone's share is in LDS

        const unsigned aoff = (kt & 1) ? (unsigned)A_BUF_BYTES : 0u;
        const unsigned boff = (kt & 1) ? (unsigned)B_BUF_BYTES : 0u;
        const unsigned short* Ab =
            (const unsigned short*)((const char*)As + aoff);
        const unsigned short* Bb =
            (const unsigned short*)((const char*)Bs + boff);

        // ---- load fragments per ISA 7.12.2 16-bit layouts ----
        FragU a[4], b[4];
#pragma unroll
        for (int f = 0; f < 4; ++f) {
            int row  = wave_m * 64 + f * 16 + lmod;
            int coff = lhalf * 8;              // lane<16 -> K0..7, lane>=16 -> K8..15
            a[f].u4[0] = *(const uint4*)(Ab + row * LDAK + coff);
            a[f].u4[1] = *(const uint4*)(Ab + row * LDAK + coff + 16);
        }
#pragma unroll
        for (int g = 0; g < 4; ++g) {
            int ncol  = wave_n * 64 + g * 16 + lmod;
            int khalf = lhalf * 8;             // lane<16 -> K0..7, lane>=16 -> K8..15
            b[g].u4[0] = *(const uint4*)(Bb + ncol * LDAK + khalf);
            b[g].u4[1] = *(const uint4*)(Bb + ncol * LDAK + khalf + 16);
        }

        // ---- kick off DMA for tile kt+1 into the other buffer ----
        if (kt + 1 < nk) {
            const unsigned an = (kt & 1) ? 0u : (unsigned)A_BUF_BYTES;
            const unsigned bn = (kt & 1) ? 0u : (unsigned)B_BUF_BYTES;
#pragma unroll
            for (int c = 0; c < 2; ++c) {
                async_copy_b128(ldsA0[c] + an, aP[c]);
                aP[c] += BK;
            }
#pragma unroll
            for (int c = 0; c < 4; ++c) {
                async_copy_b128(ldsB0[c] + bn, bP[c]);
                bP[c] += BK;
            }
        }

        // ---- 16 WMMAs per wave per K-step (overlap with DMA of tile kt+1) ----
#pragma unroll
        for (int f = 0; f < 4; ++f)
#pragma unroll
            for (int g = 0; g < 4; ++g)
                acc[f][g] = __builtin_amdgcn_wmma_f32_16x16x32_bf16(
                    false, a[f].v, false, b[g].v,
                    (short)0, acc[f][g], false, false);
    }

    // ---- epilogue: C/D 16x16 f32 layout (lane<16: M=r, lane>=16: M=8+r) ----
#pragma unroll
    for (int f = 0; f < 4; ++f) {
#pragma unroll
        for (int g = 0; g < 4; ++g) {
            int col = n0 + wave_n * 64 + g * 16 + lmod;
            float bv = has_bias ? bias[col] : 0.f;
#pragma unroll
            for (int r = 0; r < 8; ++r) {
                int row = m0 + wave_m * 64 + f * 16 + lhalf * 8 + r;
                C[(size_t)row * N + col] = acc[f][g][r] + bv;
            }
        }
    }
}

// ---------------------------------------------------------------------------
// Banded attention: one wave32 per (b, t, h). Band = keys j in [t, min(t+2,T-1)].
// ---------------------------------------------------------------------------
__global__ void __launch_bounds__(256)
banded_attention_kernel(const float* __restrict__ qkv,
                        unsigned short* __restrict__ attn) {
    const int tid  = threadIdx.x;
    const int lane = tid & 31;
    const int gw   = blockIdx.x * 8 + (tid >> 5);   // global wave id: 0..65535
    const int h    = gw & (NHEAD - 1);
    const int bt   = gw >> 4;                       // 0..4095
    const int t    = bt & (SEQ - 1);

    const int d0 = lane * 2;
    const float* q = qkv + (size_t)bt * NQKV + h * HD;
    const float q0 = q[d0], q1 = q[d0 + 1];

    float s[3];
#pragma unroll
    for (int j = 0; j < 3; ++j) {
        if (t + j < SEQ) {
            const float* kp = qkv + (size_t)(bt + j) * NQKV + DIM + h * HD;
            float p = q0 * kp[d0] + q1 * kp[d0 + 1];
#pragma unroll
            for (int off = 16; off > 0; off >>= 1)
                p += __shfl_xor(p, off, 32);
            s[j] = p * 0.125f;                      // hd^-0.5
        } else {
            s[j] = -__builtin_inff();
        }
    }

    float m = fmaxf(s[0], fmaxf(s[1], s[2]));
    float e[3];
#pragma unroll
    for (int j = 0; j < 3; ++j)
        e[j] = (t + j < SEQ) ? __expf(s[j] - m) : 0.f;
    float rinv = 1.f / (e[0] + e[1] + e[2]);

    float o0 = 0.f, o1 = 0.f;
#pragma unroll
    for (int j = 0; j < 3; ++j) {
        if (t + j < SEQ) {
            const float* vp = qkv + (size_t)(bt + j) * NQKV + 2 * DIM + h * HD;
            float pj = e[j] * rinv;
            o0 += pj * vp[d0];
            o1 += pj * vp[d0 + 1];
        }
    }
    unsigned int packed =
        ((unsigned int)f32_to_bf16(o1) << 16) | f32_to_bf16(o0);
    *(unsigned int*)(attn + (size_t)bt * DIM + h * HD + d0) = packed;
}

// ---------------------------------------------------------------------------
// Host launcher
// ---------------------------------------------------------------------------
extern "C" void kernel_launch(void* const* d_in, const int* in_sizes, int n_in,
                              void* d_out, int out_size, void* d_ws, size_t ws_size,
                              hipStream_t stream) {
    (void)in_sizes; (void)n_in; (void)out_size; (void)ws_size;
    const float* x      = (const float*)d_in[0];   // (2,2048,1024)
    const float* w_qkv  = (const float*)d_in[1];   // (1024,3072)
    const float* w_proj = (const float*)d_in[2];   // (1024,1024)
    const float* b_proj = (const float*)d_in[3];   // (1024,)

    // workspace carve-up (bytes)
    char* ws = (char*)d_ws;
    unsigned short* xb     = (unsigned short*)ws;                               //  8 MB
    unsigned short* wqkvT  = (unsigned short*)(ws + (size_t)8  * 1024 * 1024);  //  6 MB
    unsigned short* wprojT = (unsigned short*)(ws + (size_t)14 * 1024 * 1024);  //  2 MB
    float*          qkv    = (float*)         (ws + (size_t)16 * 1024 * 1024); // 48 MB
    unsigned short* attnb  = (unsigned short*)(ws + (size_t)64 * 1024 * 1024); //  8 MB

    // 1) cast x -> bf16
    {
        int n = MTOT * DIM;                       // 4,194,304
        cast_f32_bf16_kernel<<<n / (256 * 4), 256, 0, stream>>>(x, xb, n);
    }
    // 2) transpose+cast weights -> (N x K) bf16
    transpose_cast_kernel<<<dim3(NQKV / 32, DIM / 32), dim3(32, 8), 0, stream>>>(
        w_qkv, wqkvT, DIM, NQKV);
    transpose_cast_kernel<<<dim3(DIM / 32, DIM / 32), dim3(32, 8), 0, stream>>>(
        w_proj, wprojT, DIM, DIM);
    // 3) qkv = x @ Wqkv
    gemm_bf16_wmma_kernel<<<dim3(NQKV / BN, MTOT / BM), 256, 0, stream>>>(
        xb, wqkvT, qkv, nullptr, MTOT, NQKV, DIM, 0);
    // 4) banded attention -> bf16 activations
    banded_attention_kernel<<<(MTOT * NHEAD) / 8, 256, 0, stream>>>(qkv, attnb);
    // 5) out = attn @ Wproj + b
    gemm_bf16_wmma_kernel<<<dim3(DIM / BN, MTOT / BM), 256, 0, stream>>>(
        attnb, wprojT, (float*)d_out, b_proj, MTOT, DIM, DIM, 1);
}
